// TransformerBlock_17102559773000
// MI455X (gfx1250) — matrix operand order repaired and checked by reading, not compiled, not run on hardware
//
#include <hip/hip_runtime.h>
#include <hip/hip_bf16.h>
#include <math.h>

// ---------------------------------------------------------------------------
// Problem constants (B,S,H,NH from the reference)
// ---------------------------------------------------------------------------
#define BB   2
#define SS   2048
#define HH   2048
#define NHH  16
#define HD   128
#define MM   (BB * SS)        // 4096 rows of activations
#define QK_STRIDE (2 * HH)    // 4096: Q|K activation buffer row stride

typedef __attribute__((ext_vector_type(16))) _Float16 v16h;
typedef __attribute__((ext_vector_type(8)))  float    v8f;

// 16-element f16 fragment, fillable with two b128 loads
union HFrag {
  v16h  h;
  uint4 q[2];
};

// D = A(16x32 f16) * B(32x16 f16) + C(16x16 f32)  -- CDNA5 WMMA, wave32
static __device__ inline v8f wmma_f32_16x16x32_f16(v16h a, v16h b, v8f c) {
  return __builtin_amdgcn_wmma_f32_16x16x32_f16(
      /*neg_a=*/false, a, /*neg_b=*/false, b,
      /*c_mod=*/(short)0, c, /*reuse_a=*/false, /*reuse_b=*/false);
}

// Per-wave LDS producer->consumer fence. CDNA5 DS ops are in-order per wave;
// this stops compiler reordering and makes the DScnt dependency explicit.
static __device__ inline void wave_lds_fence() {
  asm volatile("s_wait_dscnt 0" ::: "memory");
  __builtin_amdgcn_wave_barrier();
}

// CDNA5 async global->LDS copy (16B per lane), tracked by ASYNCcnt.
// VDST = per-lane LDS byte address (low 32 bits of the generic __shared__
// address == workgroup LDS offset), VADDR = per-lane 64-bit global address.
static __device__ inline void async_load_b128(const void* gaddr, void* lds) {
  asm volatile("global_load_async_to_lds_b128 %0, %1, off"
               :: "v"((unsigned)(uintptr_t)lds), "v"(gaddr)
               : "memory");
}
static __device__ inline void wait_async() {
  asm volatile("s_wait_asynccnt 0" ::: "memory");
}

// ---------------------------------------------------------------------------
// f32 -> f16 conversion (weights), 4 elements/thread, vectorized
// ---------------------------------------------------------------------------
__global__ __launch_bounds__(256) void cvt_f32_f16_kernel(
    const float* __restrict__ in, _Float16* __restrict__ out, int n4) {
  int i = blockIdx.x * 256 + threadIdx.x;
  if (i < n4) {
    float4 v = ((const float4*)in)[i];
    union { _Float16 h[4]; uint2 u; } p;
    p.h[0] = (_Float16)v.x; p.h[1] = (_Float16)v.y;
    p.h[2] = (_Float16)v.z; p.h[3] = (_Float16)v.w;
    ((uint2*)out)[i] = p.u;
  }
}

// ---------------------------------------------------------------------------
// LayerNorm over H=2048, one block (256 threads) per row, f16 output
// ---------------------------------------------------------------------------
__global__ __launch_bounds__(256) void layernorm_f16_kernel(
    const float* __restrict__ x, const float* __restrict__ w,
    const float* __restrict__ b, _Float16* __restrict__ out) {
  const int row = blockIdx.x;
  const int t   = threadIdx.x;
  const float4* xr4 = (const float4*)(x + (size_t)row * HH);

  float4 va = xr4[t * 2 + 0];
  float4 vb = xr4[t * 2 + 1];
  float v[8] = {va.x, va.y, va.z, va.w, vb.x, vb.y, vb.z, vb.w};

  float s = 0.f, s2 = 0.f;
#pragma unroll
  for (int j = 0; j < 8; ++j) { s += v[j]; s2 += v[j] * v[j]; }

  __shared__ float red[512];
  red[t] = s;
  red[256 + t] = s2;
  __syncthreads();
  for (int st = 128; st > 0; st >>= 1) {
    if (t < st) {
      red[t]       += red[t + st];
      red[256 + t] += red[256 + t + st];
    }
    __syncthreads();
  }
  const float mu  = red[0] * (1.f / (float)HH);
  const float var = red[256] * (1.f / (float)HH) - mu * mu;
  const float rs  = rsqrtf(var + 1e-6f);

  const float4* w4 = (const float4*)w;
  const float4* b4 = (const float4*)b;
  float4 wa = w4[t * 2 + 0], wb = w4[t * 2 + 1];
  float4 ba = b4[t * 2 + 0], bb = b4[t * 2 + 1];
  float wv[8] = {wa.x, wa.y, wa.z, wa.w, wb.x, wb.y, wb.z, wb.w};
  float bv[8] = {ba.x, ba.y, ba.z, ba.w, bb.x, bb.y, bb.z, bb.w};

  union { _Float16 h[8]; uint4 u; } o;
#pragma unroll
  for (int j = 0; j < 8; ++j)
    o.h[j] = (_Float16)((v[j] - mu) * rs * wv[j] + bv[j]);
  *(uint4*)(out + (size_t)row * HH + t * 8) = o.u;
}

// ---------------------------------------------------------------------------
// Tiled WMMA GEMM:  C[M,N] = A[M,K] (f16 rowmajor) * W[N,K]^T (f16 rowmajor)
// 128x128 block tile, BK=32, LDS double-buffered via ASYNC global->LDS copies,
// 8 waves, wave tile 32x64. Epilogue: bias / exact GELU / f32 residual /
// f16-or-f32 out, or TRANSV: store V transposed as Vt[b][h][d][s].
// ---------------------------------------------------------------------------
template <bool BIAS, bool GELU, bool RES, bool OUT16, bool TRANSV>
__global__ __launch_bounds__(256) void gemm_wmma_kernel(
    const _Float16* __restrict__ A, const _Float16* __restrict__ W,
    const float* __restrict__ bias, const float* __restrict__ res,
    void* __restrict__ Cout, int M, int N, int K) {
  __shared__ _Float16 smem[4 * 5120];   // 2 x (128x40) A + 2 x (128x40) W

  const int t    = threadIdx.x;
  const int lane = t & 31;
  const int wave = t >> 5;
  const int wm   = wave & 3;            // 4 waves along M
  const int wn   = wave >> 2;           // 2 waves along N
  const int lo   = lane & 15;
  const int hi   = lane >> 4;
  const int bm   = blockIdx.y * 128;
  const int bn   = blockIdx.x * 128;

  v8f acc[2][4];
  {
    v8f z = {};
#pragma unroll
    for (int mf = 0; mf < 2; ++mf)
#pragma unroll
      for (int nf = 0; nf < 4; ++nf) acc[mf][nf] = z;
  }

  auto load_tile = [&](int k0, int buf) {
    _Float16* sA = smem + buf * 5120;
    _Float16* sW = smem + 10240 + buf * 5120;
#pragma unroll
    for (int c = t; c < 512; c += 256) {
      const int r  = c >> 2;
      const int kc = (c & 3) * 8;
      async_load_b128(A + (size_t)(bm + r) * K + k0 + kc, &sA[r * 40 + kc]);
      async_load_b128(W + (size_t)(bn + r) * K + k0 + kc, &sW[r * 40 + kc]);
    }
  };

  const int nk = K >> 5;
  load_tile(0, 0);
  wait_async();
  __syncthreads();
  int cur = 0;

  for (int ki = 0; ki < nk; ++ki) {
    if (ki + 1 < nk) load_tile((ki + 1) * 32, cur ^ 1);
    if (ki + 2 < nk) {
      const int pr = t & 127;
      __builtin_prefetch(A + (size_t)(bm + pr) * K + (ki + 2) * 32, 0, 1);
      __builtin_prefetch(W + (size_t)(bn + pr) * K + (ki + 2) * 32, 0, 1);
    }

    const _Float16* sA = smem + cur * 5120;
    const _Float16* sW = smem + 10240 + cur * 5120;

    HFrag af[2], bf[4];
#pragma unroll
    for (int mf = 0; mf < 2; ++mf) {
      const int row = wm * 32 + mf * 16 + lo;
      af[mf].q[0] = *(const uint4*)(&sA[row * 40 + hi * 8]);
      af[mf].q[1] = *(const uint4*)(&sA[row * 40 + 16 + hi * 8]);
    }
#pragma unroll
    for (int nf = 0; nf < 4; ++nf) {
      const int row = wn * 64 + nf * 16 + lo;
      bf[nf].q[0] = *(const uint4*)(&sW[row * 40 + hi * 8]);
      bf[nf].q[1] = *(const uint4*)(&sW[row * 40 + 16 + hi * 8]);
    }
#pragma unroll
    for (int mf = 0; mf < 2; ++mf)
#pragma unroll
      for (int nf = 0; nf < 4; ++nf)
        acc[mf][nf] = wmma_f32_16x16x32_f16(af[mf].h, bf[nf].h, acc[mf][nf]);

    wait_async();      // next tile resident in LDS
    __syncthreads();
    cur ^= 1;
  }

  if (TRANSV) {
    // Transpose the 128x128 block through LDS, store Vt[b][h][d][s].
    _Float16* tbuf = smem;   // 128 (d) x 136 (s + pad)
#pragma unroll
    for (int mf = 0; mf < 2; ++mf)
#pragma unroll
      for (int nf = 0; nf < 4; ++nf)
#pragma unroll
        for (int v = 0; v < 8; ++v) {
          const int lr = wm * 32 + mf * 16 + 8 * hi + v;   // local s
          const int lc = wn * 64 + nf * 16 + lo;           // local d
          tbuf[lc * 136 + lr] = (_Float16)acc[mf][nf][v];
        }
    __syncthreads();
    const int bb = bm >> 11;          // batch (S = 2048 rows per batch)
    const int s0 = bm & (SS - 1);
    const int hh = bn >> 7;           // head (128 cols per head)
    _Float16* vbase = (_Float16*)Cout + ((size_t)(bb * NHH + hh) * HD) * SS + s0;
    for (int c = t; c < 2048; c += 256) {
      const int d  = c >> 4;
      const int ch = (c & 15) * 8;
      *(uint4*)(vbase + (size_t)d * SS + ch) = *(const uint4*)(&tbuf[d * 136 + ch]);
    }
    return;
  }

  // Normal epilogue (C/D layout: col = lane&15, row = vgpr + 8*(lane>>4))
#pragma unroll
  for (int mf = 0; mf < 2; ++mf)
#pragma unroll
    for (int nf = 0; nf < 4; ++nf)
#pragma unroll
      for (int v = 0; v < 8; ++v) {
        const int row = bm + wm * 32 + mf * 16 + 8 * hi + v;
        const int col = bn + wn * 64 + nf * 16 + lo;
        float xv = acc[mf][nf][v];
        if (BIAS) xv += bias[col];
        if (GELU) xv = 0.5f * xv * (1.f + erff(xv * 0.70710678118654752f));
        if (RES)  xv += res[(size_t)row * N + col];
        if (OUT16)
          ((_Float16*)Cout)[(size_t)row * N + col] = (_Float16)xv;
        else
          ((float*)Cout)[(size_t)row * N + col] = xv;
      }
}

// ---------------------------------------------------------------------------
// Flash attention (causal), block-cooperative K/V staging via async copies.
//   qk : [B*S, 2H] f16 rowmajor, columns = Q (head-major) | K (head-major)
//   vt : [B][NH][HD][S] f16  (transposed V -> contiguous along keys)
// Grid (S/128, B*NH), 8 waves/block covering 128 query rows. Each 32-key
// block of K (32x128, [key][d]) and V (128x32, [d][key]) is staged once in
// LDS and consumed by all 8 waves; the k-loop trip count is block-uniform so
// __syncthreads is legal, and waves past their causal limit skip compute.
// ---------------------------------------------------------------------------
__global__ __launch_bounds__(256) void flash_attn_kernel(
    const _Float16* __restrict__ qk, const _Float16* __restrict__ vt,
    _Float16* __restrict__ out) {
  const int t    = threadIdx.x;
  const int lane = t & 31;
  const int wave = t >> 5;
  const int b    = blockIdx.y >> 4;
  const int h    = blockIdx.y & 15;
  const int q0   = blockIdx.x * 128;
  const int qw   = q0 + wave * 16;   // first query row of this wave
  const int lo   = lane & 15;
  const int hi   = lane >> 4;

  __shared__ _Float16 sK[32 * 136];        // [key][d],  pitch 136 (8.5 KB)
  __shared__ _Float16 sV[128 * 40];        // [d][key],  pitch 40  (10 KB)
  __shared__ float    sScore[8][16][36];   // per-wave, 16B rows, no conflicts
  __shared__ _Float16 sP[8][16][40];
  __shared__ float    sFac[8][16];
  float    (*score)[36] = sScore[wave];
  _Float16 (*P)[40]     = sP[wave];
  float*   fac          = sFac[wave];

  const _Float16* Qb = qk + (size_t)(b * SS) * QK_STRIDE + h * HD;
  const _Float16* Kg = Qb + HH;
  const _Float16* Vt = vt + ((size_t)(b * NHH + h) * HD) * SS;

  // Q fragments for this wave's 16 rows: 4 K-steps of 32 over HD=128
  HFrag qf[4];
  {
    const _Float16* qrow = Qb + (size_t)(qw + lo) * QK_STRIDE;
#pragma unroll
    for (int ks = 0; ks < 4; ++ks) {
      qf[ks].q[0] = *(const uint4*)(qrow + ks * 32 + hi * 8);
      qf[ks].q[1] = *(const uint4*)(qrow + ks * 32 + 16 + hi * 8);
    }
  }

  v8f acc[8];   // 16 x 128 f32 output accumulator (8 N-frags over head dim)
  {
    v8f z = {};
#pragma unroll
    for (int nf = 0; nf < 8; ++nf) acc[nf] = z;
  }

  float m_i = -INFINITY, l_i = 0.f;
  const float scale = 0.08838834764831845f;   // 1/sqrt(128)

  const int kend = q0 + 127;   // block-uniform causal bound
  for (int kb = 0; kb <= kend; kb += 32) {
    __syncthreads();   // previous block's LDS reads complete
    // ---- cooperative async staging: K tile (32x128) and V tile (128x32) ----
    {
      int c = t;
#pragma unroll
      for (int it = 0; it < 2; ++it, c += 256) {
        const int key = c >> 4;
        const int ch  = (c & 15) * 8;
        async_load_b128(Kg + (size_t)(kb + key) * QK_STRIDE + ch,
                        &sK[key * 136 + ch]);
        const int d   = c >> 2;
        const int chk = (c & 3) * 8;
        async_load_b128(Vt + (size_t)d * SS + kb + chk,
                        &sV[d * 40 + chk]);
      }
      if (kb + 32 <= kend) {
        const int key = t >> 4, d = t >> 1;
        __builtin_prefetch(Kg + (size_t)(kb + 32 + key) * QK_STRIDE + (t & 15) * 8, 0, 1);
        __builtin_prefetch(Vt + (size_t)d * SS + kb + 32 + (t & 1) * 16, 0, 1);
      }
    }
    wait_async();
    __syncthreads();   // staging visible to all waves

    if (kb > qw + 15) continue;   // beyond this wave's causal limit

    // ---- S = Q * K^T for 32 keys (2 N-frags, 4 K-steps each) ----
#pragma unroll
    for (int nb = 0; nb < 2; ++nb) {
      v8f sc = {};
      const _Float16* krow = &sK[(nb * 16 + lo) * 136];
#pragma unroll
      for (int ks = 0; ks < 4; ++ks) {
        HFrag bf;
        bf.q[0] = *(const uint4*)(krow + ks * 32 + hi * 8);
        bf.q[1] = *(const uint4*)(krow + ks * 32 + 16 + hi * 8);
        sc = wmma_f32_16x16x32_f16(qf[ks].h, bf.h, sc);
      }
#pragma unroll
      for (int v = 0; v < 8; ++v)
        score[v + 8 * hi][nb * 16 + lo] = sc[v] * scale;
    }
    wave_lds_fence();

    // ---- online softmax, one query row per lane (lanes 0..15) ----
    if (lane < 16) {
      const int r = lane;
      int lim = qw + r + 1 - kb;          // causal: keys kb..kb+lim-1 valid
      if (lim > 32) lim = 32;
      float sv[32];
      const float4* srow = (const float4*)score[r];
#pragma unroll
      for (int jq = 0; jq < 8; ++jq) {
        float4 f4 = srow[jq];
        sv[jq * 4 + 0] = f4.x; sv[jq * 4 + 1] = f4.y;
        sv[jq * 4 + 2] = f4.z; sv[jq * 4 + 3] = f4.w;
      }
      float mnew = m_i;
      for (int j = 0; j < lim; ++j) mnew = fmaxf(mnew, sv[j]);
      const float f = __expf(m_i - mnew);   // first block: exp(-inf)=0
      float sum = l_i * f;
#pragma unroll
      for (int jq = 0; jq < 8; ++jq) {
        union { _Float16 hh[4]; uint2 u; } pk;
#pragma unroll
        for (int e = 0; e < 4; ++e) {
          const int j = jq * 4 + e;
          const float p = (j < lim) ? __expf(sv[j] - mnew) : 0.f;
          sum += p;
          pk.hh[e] = (_Float16)p;
        }
        *(uint2*)(&P[r][jq * 4]) = pk.u;
      }
      m_i = mnew;
      l_i = sum;
      fac[r] = f;
    }
    wave_lds_fence();

    // ---- rescale accumulator by per-row factor ----
    float fr[8];
#pragma unroll
    for (int v = 0; v < 8; ++v) fr[v] = fac[8 * hi + v];
#pragma unroll
    for (int nf = 0; nf < 8; ++nf)
#pragma unroll
      for (int v = 0; v < 8; ++v) acc[nf][v] *= fr[v];

    // ---- O += P(16x32) * V(32x128), V fragments from LDS [d][key] ----
    HFrag pf;
    pf.q[0] = *(const uint4*)(&P[lo][hi * 8]);
    pf.q[1] = *(const uint4*)(&P[lo][16 + hi * 8]);
#pragma unroll
    for (int nf = 0; nf < 8; ++nf) {
      const _Float16* vrow = &sV[(nf * 16 + lo) * 40 + hi * 8];
      HFrag vf;
      vf.q[0] = *(const uint4*)(vrow);
      vf.q[1] = *(const uint4*)(vrow + 16);
      acc[nf] = wmma_f32_16x16x32_f16(pf.h, vf.h, acc[nf]);
    }
  }

  // ---- normalize by row sum and store f16 ----
  if (lane < 16) fac[lane] = 1.f / l_i;
  wave_lds_fence();
  float fr[8];
#pragma unroll
  for (int v = 0; v < 8; ++v) fr[v] = fac[8 * hi + v];
#pragma unroll
  for (int nf = 0; nf < 8; ++nf)
#pragma unroll
    for (int v = 0; v < 8; ++v) {
      const int r = 8 * hi + v;
      out[(size_t)(b * SS + qw + r) * HH + h * HD + nf * 16 + lo] =
          (_Float16)(acc[nf][v] * fr[v]);
    }
}

// ---------------------------------------------------------------------------
// Host orchestration
// ---------------------------------------------------------------------------
extern "C" void kernel_launch(void* const* d_in, const int* in_sizes, int n_in,
                              void* d_out, int out_size, void* d_ws, size_t ws_size,
                              hipStream_t stream) {
  (void)in_sizes; (void)n_in; (void)out_size; (void)ws_size;

  const float* x     = (const float*)d_in[0];
  const float* ln1w  = (const float*)d_in[1];
  const float* ln1b  = (const float*)d_in[2];
  const float* ln2w  = (const float*)d_in[3];
  const float* ln2b  = (const float*)d_in[4];
  const float* qkv_w = (const float*)d_in[5];
  const float* out_w = (const float*)d_in[6];
  const float* fc1_w = (const float*)d_in[7];
  const float* fc1_b = (const float*)d_in[8];
  const float* fc2_w = (const float*)d_in[9];
  const float* fc2_b = (const float*)d_in[10];
  // d_in[11] = attention_mask: causal, applied analytically in-kernel.

  char* ws = (char*)d_ws;
  size_t off = 0;
  auto alloc = [&](size_t bytes) -> char* {
    char* p = ws + off;
    off += (bytes + 255) & ~(size_t)255;
    return p;
  };

  _Float16* wq16  = (_Float16*)alloc((size_t)3 * HH * HH * 2);   // qkv_w f16
  _Float16* wo16  = (_Float16*)alloc((size_t)HH * HH * 2);
  _Float16* wf1   = (_Float16*)alloc((size_t)4 * HH * HH * 2);
  _Float16* wf2   = (_Float16*)alloc((size_t)4 * HH * HH * 2);
  _Float16* h1    = (_Float16*)alloc((size_t)MM * HH * 2);
  _Float16* qk16  = (_Float16*)alloc((size_t)MM * 2 * HH * 2);   // Q|K rowmajor
  _Float16* vt16  = (_Float16*)alloc((size_t)MM * HH * 2);       // V transposed
  _Float16* att16 = (_Float16*)alloc((size_t)MM * HH * 2);
  float*    x1    = (float*)   alloc((size_t)MM * HH * 4);
  _Float16* h2    = (_Float16*)alloc((size_t)MM * HH * 2);
  _Float16* fc1a  = (_Float16*)alloc((size_t)MM * 4 * HH * 2);

  // 1) weights -> f16 (vectorized x4)
  cvt_f32_f16_kernel<<<(3 * HH * HH / 4 + 255) / 256, 256, 0, stream>>>(qkv_w, wq16, 3 * HH * HH / 4);
  cvt_f32_f16_kernel<<<(HH * HH / 4 + 255) / 256, 256, 0, stream>>>(out_w, wo16, HH * HH / 4);
  cvt_f32_f16_kernel<<<(4 * HH * HH / 4 + 255) / 256, 256, 0, stream>>>(fc1_w, wf1, 4 * HH * HH / 4);
  cvt_f32_f16_kernel<<<(4 * HH * HH / 4 + 255) / 256, 256, 0, stream>>>(fc2_w, wf2, 4 * HH * HH / 4);

  // 2) h1 = LN1(x) in f16
  layernorm_f16_kernel<<<MM, 256, 0, stream>>>(x, ln1w, ln1b, h1);

  // 3a) qk = h1 @ [Wq;Wk]^T   [4096 x 4096] rowmajor
  gemm_wmma_kernel<false, false, false, true, false>
      <<<dim3(2 * HH / 128, MM / 128), 256, 0, stream>>>(
      h1, wq16, nullptr, nullptr, qk16, MM, 2 * HH, HH);

  // 3b) Vt = (h1 @ Wv^T) transposed -> [B][NH][HD][S]
  gemm_wmma_kernel<false, false, false, true, true>
      <<<dim3(HH / 128, MM / 128), 256, 0, stream>>>(
      h1, wq16 + (size_t)2 * HH * HH, nullptr, nullptr, vt16, MM, HH, HH);

  // 4) flash attention -> att16 [4096 x 2048]
  flash_attn_kernel<<<dim3(SS / 128, BB * NHH), 256, 0, stream>>>(qk16, vt16, att16);

  // 5) x1 = x + att16 @ out_w^T   (f32)
  gemm_wmma_kernel<false, false, true, false, false>
      <<<dim3(HH / 128, MM / 128), 256, 0, stream>>>(
      att16, wo16, nullptr, x, x1, MM, HH, HH);

  // 6) h2 = LN2(x1) in f16
  layernorm_f16_kernel<<<MM, 256, 0, stream>>>(x1, ln2w, ln2b, h2);

  // 7) fc1a = gelu(h2 @ fc1_w^T + fc1_b)   [4096 x 8192] f16
  gemm_wmma_kernel<true, true, false, true, false>
      <<<dim3(4 * HH / 128, MM / 128), 256, 0, stream>>>(
      h2, wf1, fc1_b, nullptr, fc1a, MM, 4 * HH, HH);

  // 8) out = x1 + fc1a @ fc2_w^T + fc2_b   (f32 -> d_out)
  gemm_wmma_kernel<true, false, true, false, false>
      <<<dim3(HH / 128, MM / 128), 256, 0, stream>>>(
      fc1a, wf2, fc2_b, x1, (float*)d_out, MM, HH, 4 * HH);
}
